// MultiHeadAttention_38628935860377
// MI455X (gfx1250) — matrix operand order repaired and checked by reading, not compiled, not run on hardware
//
#include <hip/hip_runtime.h>
#include <hip/hip_bf16.h>

// ---------------------------------------------------------------------------
// MI455X (gfx1250) fused pre-LN multi-head attention block.
// Pipeline: LN(f32->f16) -> WMMA QKV GEMM (scatter to head-major Q/K/V f16)
//           -> RoPE in-place -> flash-attention (64-key blocks, WMMA + online
//           softmax, async-to-LDS K staging) -> WMMA output projection.
// GEMMs are register-blocked 2x2 WMMA tiles per wave (32x32 out per wave).
// ---------------------------------------------------------------------------

#define D_MODEL   1024
#define NUM_HEADS 16
#define HEAD_DIM  64
#define SEQ       2048
#define BATCH     4

typedef __attribute__((ext_vector_type(16))) _Float16 v16h;
typedef __attribute__((ext_vector_type(8)))  _Float16 v8h;
typedef __attribute__((ext_vector_type(4)))  _Float16 v4h;
typedef __attribute__((ext_vector_type(8)))  float    v8f;

union V16U { v16h v; v8h h[2]; };

__device__ __forceinline__ v8f wmma_f16(v16h a, v16h b, v8f c) {
  return __builtin_amdgcn_wmma_f32_16x16x32_f16(false, a, false, b, (short)0, c,
                                                false, false);
}

// Assemble a v16h from two 16B-aligned v8h loads (works for LDS and global).
__device__ __forceinline__ v16h ld16(const _Float16* p) {
  V16U u;
  u.h[0] = *(const v8h*)p;
  u.h[1] = *(const v8h*)(p + 8);
  return u.v;
}

// Async global->LDS 16B copy (per lane), tracked by ASYNCcnt.
__device__ __forceinline__ void async_g2l_b128(const void* gptr, void* lptr) {
  const unsigned lds = (unsigned)(unsigned long long)lptr;  // LDS = addr[31:0]
  const unsigned long long ga = (unsigned long long)gptr;
  asm volatile("global_load_async_to_lds_b128 %0, %1, off"
               :: "v"(lds), "v"(ga) : "memory");
}
__device__ __forceinline__ void wait_asynccnt0() {
  asm volatile("s_wait_asynccnt 0x0" ::: "memory");
}

// ---------------------------------------------------------------------------
// f32 -> f16 weight conversion
// ---------------------------------------------------------------------------
__global__ void cvt_f32_f16_kernel(const float* __restrict__ src,
                                   _Float16* __restrict__ dst, int n) {
  for (int i = blockIdx.x * blockDim.x + threadIdx.x; i < n;
       i += gridDim.x * blockDim.x)
    dst[i] = (_Float16)src[i];
}

// ---------------------------------------------------------------------------
// LayerNorm (gamma only) -> f16. One block per row.
// ---------------------------------------------------------------------------
__global__ __launch_bounds__(256) void ln_kernel(const float* __restrict__ x,
                                                 const float* __restrict__ gamma,
                                                 _Float16* __restrict__ xn) {
  const int row = blockIdx.x;
  const int tid = threadIdx.x;
  const float* xr = x + (size_t)row * D_MODEL;
  float4 v = ((const float4*)xr)[tid];  // 256 threads * 4 = 1024
  float s  = v.x + v.y + v.z + v.w;
  float s2 = v.x * v.x + v.y * v.y + v.z * v.z + v.w * v.w;
#pragma unroll
  for (int off = 1; off < 32; off <<= 1) {
    s  += __shfl_xor(s,  off, 32);
    s2 += __shfl_xor(s2, off, 32);
  }
  __shared__ float ws1[8], ws2[8];
  if ((tid & 31) == 0) { ws1[tid >> 5] = s; ws2[tid >> 5] = s2; }
  __syncthreads();
  float ts = 0.f, ts2 = 0.f;
#pragma unroll
  for (int i = 0; i < 8; ++i) { ts += ws1[i]; ts2 += ws2[i]; }
  const float mu   = ts * (1.0f / D_MODEL);
  const float var  = ts2 * (1.0f / D_MODEL) - mu * mu;
  const float rinv = rsqrtf(var + 1e-5f);
  float4 g = ((const float4*)gamma)[tid];
  v4h h;
  h[0] = (_Float16)((v.x - mu) * rinv * g.x);
  h[1] = (_Float16)((v.y - mu) * rinv * g.y);
  h[2] = (_Float16)((v.z - mu) * rinv * g.z);
  h[3] = (_Float16)((v.w - mu) * rinv * g.w);
  *(v4h*)(xn + (size_t)row * D_MODEL + tid * 4) = h;
}

// ---------------------------------------------------------------------------
// Epilogue store for one 16x16 C tile (8 strided elements per lane).
// EPI 0: f32 row-major [M,N] into C0.
// EPI 1: QKV scatter: col = which*1024 + h*64 + dh -> f16 [B][H][S][64].
// ---------------------------------------------------------------------------
template <int EPI>
__device__ __forceinline__ void epi_store(const v8f& acc, int rbase, int col,
                                          int N, void* C0, void* C1, void* C2) {
  if (EPI == 0) {
    float* C = (float*)C0;
#pragma unroll
    for (int r = 0; r < 8; ++r)
      C[(size_t)(rbase + r) * N + col] = acc[r];
  } else {
    const int which = col >> 10;          // 0=q 1=k 2=v
    const int hd    = col & 1023;         // h*64 + dh
    const int h     = hd >> 6;
    const int dh    = hd & 63;
    _Float16* base = (which == 0) ? (_Float16*)C0
                   : (which == 1) ? (_Float16*)C1
                                  : (_Float16*)C2;
#pragma unroll
    for (int r = 0; r < 8; ++r) {
      const int row = rbase + r;
      const int bb  = row >> 11;          // / SEQ
      const int ss  = row & (SEQ - 1);
      base[(((size_t)bb * NUM_HEADS + h) * SEQ + ss) * HEAD_DIM + dh] =
          (_Float16)acc[r];
    }
  }
}

// ---------------------------------------------------------------------------
// Tiled WMMA GEMM: C[M,N] = A[M,K] * B[K,N], f16 in, f32 accum.
// Block = 256 threads (8 waves), tile 128(M) x 64(N), k-step 32.
// Each wave register-blocks a 32x32 output = 2x2 WMMA tiles: 4 WMMAs per
// k-step on 4 fragment loads (A/B each reused twice).
// ---------------------------------------------------------------------------
template <int EPI>
__global__ __launch_bounds__(256) void gemm_wmma_kernel(
    const _Float16* __restrict__ A, const _Float16* __restrict__ Bw,
    void* __restrict__ C0, void* __restrict__ C1, void* __restrict__ C2,
    int M, int N, int K) {
  __shared__ _Float16 Bt[64][40];  // [n][k] transposed, 80B rows (16B aligned)

  const int tid    = threadIdx.x;
  const int lane   = tid & 31;
  const int w      = tid >> 5;
  const int lane16 = lane & 15;
  const int hi     = (lane >> 4) & 1;
  const int mi     = w >> 1;     // 0..3
  const int ni     = w & 1;      // 0..1
  const int m0 = blockIdx.y * 128;
  const int n0 = blockIdx.x * 64;

  const _Float16* Ap0 = A + (size_t)(m0 + mi * 32 + lane16) * K;
  const _Float16* Ap1 = Ap0 + (size_t)16 * K;
  const int aoff_lo = hi * 16;       // A: halves 0..7  = K {0..7 | 16..23}
  const int aoff_hi = hi * 16 + 8;   //    halves 8..15 = K {8..15|24..31}

  const int bkk = tid >> 3;          // 0..31 contraction index within tile
  const int bn8 = (tid & 7) * 8;     // 0..56 n within tile

  v8f acc00 = {}, acc01 = {}, acc10 = {}, acc11 = {};
  for (int kk = 0; kk < K; kk += 32) {
    __syncthreads();
    // cooperative transposed B tile: Bt[n][k] = Bw[kk+k][n0+n]
    v8h bc = *(const v8h*)(Bw + (size_t)(kk + bkk) * N + n0 + bn8);
#pragma unroll
    for (int i = 0; i < 8; ++i) Bt[bn8 + i][bkk] = bc[i];
    if (kk + 32 < K)
      __builtin_prefetch(Bw + (size_t)(kk + 32 + bkk) * N + n0 + bn8, 0, 1);
    __syncthreads();

    V16U a0, a1;
    a0.h[0] = *(const v8h*)(Ap0 + kk + aoff_lo);
    a0.h[1] = *(const v8h*)(Ap0 + kk + aoff_hi);
    a1.h[0] = *(const v8h*)(Ap1 + kk + aoff_lo);
    a1.h[1] = *(const v8h*)(Ap1 + kk + aoff_hi);
    v16h b0 = ld16(&Bt[ni * 32 + lane16][hi * 16]);
    v16h b1 = ld16(&Bt[ni * 32 + 16 + lane16][hi * 16]);
    acc00 = wmma_f16(a0.v, b0, acc00);
    acc01 = wmma_f16(a0.v, b1, acc01);
    acc10 = wmma_f16(a1.v, b0, acc10);
    acc11 = wmma_f16(a1.v, b1, acc11);
  }

  const int col0 = n0 + ni * 32 + lane16;
  const int rb0  = m0 + mi * 32 + hi * 8;
  epi_store<EPI>(acc00, rb0,      col0,      N, C0, C1, C2);
  epi_store<EPI>(acc01, rb0,      col0 + 16, N, C0, C1, C2);
  epi_store<EPI>(acc10, rb0 + 16, col0,      N, C0, C1, C2);
  epi_store<EPI>(acc11, rb0 + 16, col0 + 16, N, C0, C1, C2);
}

// ---------------------------------------------------------------------------
// RoPE in place on Q and K ([B][H][S][64] f16). Each thread owns one
// (dh, dh+32) pair -> no races. Folds attention scale 1/8 into Q.
// ---------------------------------------------------------------------------
__global__ __launch_bounds__(256) void rope_kernel(_Float16* __restrict__ Qh,
                                                   _Float16* __restrict__ Kh,
                                                   const float* __restrict__ cosb,
                                                   const float* __restrict__ sinb) {
  const int gid   = blockIdx.x * 256 + threadIdx.x;
  const int j     = gid & 31;
  const int rowid = gid >> 5;             // [0, B*H*S)
  const int s     = rowid & (SEQ - 1);
  _Float16* q = Qh + (size_t)rowid * HEAD_DIM;
  _Float16* k = Kh + (size_t)rowid * HEAD_DIM;
  const float c0 = cosb[s * HEAD_DIM + j];
  const float s0 = sinb[s * HEAD_DIM + j];
  const float c1 = cosb[s * HEAD_DIM + j + 32];
  const float s1 = sinb[s * HEAD_DIM + j + 32];
  float q0 = (float)q[j], q1 = (float)q[j + 32];
  float k0 = (float)k[j], k1 = (float)k[j + 32];
  q[j]      = (_Float16)((q0 * c0 - q1 * s0) * 0.125f);
  q[j + 32] = (_Float16)((q1 * c1 + q0 * s1) * 0.125f);
  k[j]      = (_Float16)(k0 * c0 - k1 * s0);
  k[j + 32] = (_Float16)(k1 * c1 + k0 * s1);
}

// ---------------------------------------------------------------------------
// Flash attention, 64-key blocks. Block = (b, h, 128 q rows); 8 waves * 16
// q rows each. K tile staged with async global->LDS (ASYNCcnt path); V tile
// transposed manually. 16 WMMAs per key block per wave; softmax shfl_xor
// reductions amortized over 64 keys.
// Out: f16 [B][S][H*64] ready for the output projection GEMM.
// ---------------------------------------------------------------------------
__global__ __launch_bounds__(256) void attn_kernel(
    const _Float16* __restrict__ Qh, const _Float16* __restrict__ Kh,
    const _Float16* __restrict__ Vh, const float* __restrict__ mask,
    _Float16* __restrict__ Out) {
  __shared__ _Float16 Kt[64][64];      // [key][dh]                     8 KB
  __shared__ _Float16 Vt[64][72];      // [dh][key] transposed, padded  9 KB
  __shared__ _Float16 Pw[8][16][72];   // per-wave P scratch, padded   18 KB

  const int tid    = threadIdx.x;
  const int lane   = tid & 31;
  const int w      = tid >> 5;
  const int lane16 = lane & 15;
  const int hi     = (lane >> 4) & 1;
  const int b  = blockIdx.z;
  const int h  = blockIdx.y;
  const int q0 = blockIdx.x * 128;

  const size_t headoff = ((size_t)b * NUM_HEADS + h) * SEQ * HEAD_DIM;

  // Q A-fragments for the two 32-wide contraction chunks of Dh=64.
  const _Float16* Qp = Qh + headoff + (size_t)(q0 + w * 16 + lane16) * HEAD_DIM;
  V16U aq0, aq1;
  aq0.h[0] = *(const v8h*)(Qp + hi * 16);
  aq0.h[1] = *(const v8h*)(Qp + 8 + hi * 16);
  aq1.h[0] = *(const v8h*)(Qp + 32 + hi * 16);
  aq1.h[1] = *(const v8h*)(Qp + 40 + hi * 16);

  v8f o0 = {}, o1 = {}, o2 = {}, o3 = {};
  float mrow[8], lrow[8];
#pragma unroll
  for (int r = 0; r < 8; ++r) { mrow[r] = -3.0e38f; lrow[r] = 0.f; }

  const int ckey = tid >> 2;         // key within tile (0..63)
  const int cd16 = (tid & 3) * 16;   // dh chunk base
  const float* maskp = mask + (size_t)b * SEQ * SEQ;
  const int qr = q0 + w * 16 + hi * 8;

  for (int kb = 0; kb < SEQ; kb += 64) {
    __syncthreads();
    // K tile: async global->LDS, 2 x 16B per thread (ASYNCcnt tracked)
    {
      const _Float16* Kg = Kh + headoff + (size_t)(kb + ckey) * HEAD_DIM + cd16;
      async_g2l_b128(Kg,     &Kt[ckey][cd16]);
      async_g2l_b128(Kg + 8, &Kt[ckey][cd16 + 8]);
    }
    // V tile: manual transposed staging
    {
      const _Float16* Vg = Vh + headoff + (size_t)(kb + ckey) * HEAD_DIM + cd16;
      v8h v0 = *(const v8h*)Vg;
      v8h v1 = *(const v8h*)(Vg + 8);
#pragma unroll
      for (int i = 0; i < 8; ++i) {
        Vt[cd16 + i][ckey]     = v0[i];
        Vt[cd16 + 8 + i][ckey] = v1[i];
      }
    }
    wait_asynccnt0();
    __syncthreads();

    // logits: four 16-key tiles, contraction Dh=64 in two WMMA steps each
    v8f s0 = {}, s1 = {}, s2 = {}, s3 = {};
    s0 = wmma_f16(aq0.v, ld16(&Kt[lane16][hi * 16]), s0);
    s0 = wmma_f16(aq1.v, ld16(&Kt[lane16][32 + hi * 16]), s0);
    s1 = wmma_f16(aq0.v, ld16(&Kt[16 + lane16][hi * 16]), s1);
    s1 = wmma_f16(aq1.v, ld16(&Kt[16 + lane16][32 + hi * 16]), s1);
    s2 = wmma_f16(aq0.v, ld16(&Kt[32 + lane16][hi * 16]), s2);
    s2 = wmma_f16(aq1.v, ld16(&Kt[32 + lane16][32 + hi * 16]), s2);
    s3 = wmma_f16(aq0.v, ld16(&Kt[48 + lane16][hi * 16]), s3);
    s3 = wmma_f16(aq1.v, ld16(&Kt[48 + lane16][32 + hi * 16]), s3);

    // additive mask
#pragma unroll
    for (int r = 0; r < 8; ++r) {
      const float* mp = maskp + (size_t)(qr + r) * SEQ + kb;
      s0[r] += mp[lane16];
      s1[r] += mp[16 + lane16];
      s2[r] += mp[32 + lane16];
      s3[r] += mp[48 + lane16];
    }

    // online softmax over 64 keys (exp in place: s -> p)
#pragma unroll
    for (int r = 0; r < 8; ++r) {
      float t = fmaxf(fmaxf(s0[r], s1[r]), fmaxf(s2[r], s3[r]));
      t = fmaxf(t, __shfl_xor(t, 1, 32));
      t = fmaxf(t, __shfl_xor(t, 2, 32));
      t = fmaxf(t, __shfl_xor(t, 4, 32));
      t = fmaxf(t, __shfl_xor(t, 8, 32));
      const float nm  = fmaxf(mrow[r], t);
      const float scl = __expf(mrow[r] - nm);
      mrow[r] = nm;
      s0[r] = __expf(s0[r] - nm);
      s1[r] = __expf(s1[r] - nm);
      s2[r] = __expf(s2[r] - nm);
      s3[r] = __expf(s3[r] - nm);
      float rs = (s0[r] + s1[r]) + (s2[r] + s3[r]);
      rs += __shfl_xor(rs, 1, 32);
      rs += __shfl_xor(rs, 2, 32);
      rs += __shfl_xor(rs, 4, 32);
      rs += __shfl_xor(rs, 8, 32);
      lrow[r] = lrow[r] * scl + rs;
      o0[r] *= scl; o1[r] *= scl; o2[r] *= scl; o3[r] *= scl;
    }

    // C-layout -> A-layout for P via per-wave LDS scratch
#pragma unroll
    for (int r = 0; r < 8; ++r) {
      Pw[w][hi * 8 + r][lane16]      = (_Float16)s0[r];
      Pw[w][hi * 8 + r][16 + lane16] = (_Float16)s1[r];
      Pw[w][hi * 8 + r][32 + lane16] = (_Float16)s2[r];
      Pw[w][hi * 8 + r][48 + lane16] = (_Float16)s3[r];
    }
    V16U pa0, pa1;  // two 16x32 A fragments covering keys 0..31 / 32..63
    pa0.h[0] = *(const v8h*)&Pw[w][lane16][hi * 16];
    pa0.h[1] = *(const v8h*)&Pw[w][lane16][8 + hi * 16];
    pa1.h[0] = *(const v8h*)&Pw[w][lane16][32 + hi * 16];
    pa1.h[1] = *(const v8h*)&Pw[w][lane16][40 + hi * 16];

    // P * V : 4 dh tiles x 2 key chunks
    o0 = wmma_f16(pa0.v, ld16(&Vt[lane16][hi * 16]), o0);
    o0 = wmma_f16(pa1.v, ld16(&Vt[lane16][32 + hi * 16]), o0);
    o1 = wmma_f16(pa0.v, ld16(&Vt[16 + lane16][hi * 16]), o1);
    o1 = wmma_f16(pa1.v, ld16(&Vt[16 + lane16][32 + hi * 16]), o1);
    o2 = wmma_f16(pa0.v, ld16(&Vt[32 + lane16][hi * 16]), o2);
    o2 = wmma_f16(pa1.v, ld16(&Vt[32 + lane16][32 + hi * 16]), o2);
    o3 = wmma_f16(pa0.v, ld16(&Vt[48 + lane16][hi * 16]), o3);
    o3 = wmma_f16(pa1.v, ld16(&Vt[48 + lane16][32 + hi * 16]), o3);
  }

  // finalize + write f16 [B][S][H*64]
#pragma unroll
  for (int r = 0; r < 8; ++r) {
    const float inv = 1.0f / lrow[r];
    const size_t rowoff = ((size_t)b * SEQ + qr + r) * D_MODEL + h * HEAD_DIM;
    Out[rowoff + 0  + lane16] = (_Float16)(o0[r] * inv);
    Out[rowoff + 16 + lane16] = (_Float16)(o1[r] * inv);
    Out[rowoff + 32 + lane16] = (_Float16)(o2[r] * inv);
    Out[rowoff + 48 + lane16] = (_Float16)(o3[r] * inv);
  }
}

// ---------------------------------------------------------------------------
// Launch
// ---------------------------------------------------------------------------
extern "C" void kernel_launch(void* const* d_in, const int* in_sizes, int n_in,
                              void* d_out, int out_size, void* d_ws, size_t ws_size,
                              hipStream_t stream) {
  const float* inputs = (const float*)d_in[0];  // [B,S,D]
  const float* mask   = (const float*)d_in[1];  // [B,1,S,S]
  const float* cosb   = (const float*)d_in[2];  // [S,64]
  const float* sinb   = (const float*)d_in[3];  // [S,64]
  const float* gamma  = (const float*)d_in[4];  // [D]
  const float* w_qkv  = (const float*)d_in[5];  // [D,3D]
  const float* w_o    = (const float*)d_in[6];  // [D,D]
  float* out = (float*)d_out;

  const int M = BATCH * SEQ;  // 8192

  // workspace carve-up (f16 buffers)
  char* ws = (char*)d_ws;
  size_t off = 0;
  auto carve = [&](size_t bytes) {
    char* p = ws + off;
    off += (bytes + 255) & ~(size_t)255;
    return p;
  };
  _Float16* xn    = (_Float16*)carve((size_t)M * D_MODEL * 2);
  _Float16* wqkvh = (_Float16*)carve((size_t)D_MODEL * 3 * D_MODEL * 2);
  _Float16* woh   = (_Float16*)carve((size_t)D_MODEL * D_MODEL * 2);
  _Float16* Qhd   = (_Float16*)carve((size_t)M * D_MODEL * 2);
  _Float16* Khd   = (_Float16*)carve((size_t)M * D_MODEL * 2);
  _Float16* Vhd   = (_Float16*)carve((size_t)M * D_MODEL * 2);
  _Float16* attn_out = xn;  // xn is dead after the QKV GEMM — reuse

  // 1) weights -> f16
  cvt_f32_f16_kernel<<<768, 256, 0, stream>>>(w_qkv, wqkvh, D_MODEL * 3 * D_MODEL);
  cvt_f32_f16_kernel<<<768, 256, 0, stream>>>(w_o, woh, D_MODEL * D_MODEL);

  // 2) LayerNorm -> f16
  ln_kernel<<<M, 256, 0, stream>>>(inputs, gamma, xn);

  // 3) QKV GEMM with head-major scatter epilogue
  gemm_wmma_kernel<1><<<dim3(3 * D_MODEL / 64, M / 128), 256, 0, stream>>>(
      xn, wqkvh, Qhd, Khd, Vhd, M, 3 * D_MODEL, D_MODEL);

  // 4) RoPE in place (scale 1/8 folded into Q)
  rope_kernel<<<(BATCH * NUM_HEADS * SEQ * 32) / 256, 256, 0, stream>>>(
      Qhd, Khd, cosb, sinb);

  // 5) flash attention
  attn_kernel<<<dim3(SEQ / 128, NUM_HEADS, BATCH), 256, 0, stream>>>(
      Qhd, Khd, Vhd, mask, attn_out);

  // 6) output projection -> f32
  gemm_wmma_kernel<0><<<dim3(D_MODEL / 64, M / 128), 256, 0, stream>>>(
      attn_out, woh, out, nullptr, nullptr, M, D_MODEL, D_MODEL);
}
